// BatchedMHC_39831526703216
// MI455X (gfx1250) — matrix operand order repaired and checked by reading, not compiled, not run on hardware
//
#include <hip/hip_runtime.h>
#include <hip/hip_bf16.h>

// ---------------- problem constants ----------------
#define NEXP   64
#define DH     512
#define NS     4
#define NDH    2048        // NS*DH
#define KACT   8
#define BB     8
#define TT     4096
#define NTOK   (BB*TT)     // 32768 tokens
#define NROWS  (KACT*24)   // 192 fused weight rows
#define MBLK   32          // tokens per workgroup
#define XS_STRIDE 4112     // bytes/token row in LDS: 2048*2 + 16 pad (bank-conflict-free b128)
#define ZS_STRIDE 784      // bytes/token row for z: 196 floats (padded)

typedef __attribute__((ext_vector_type(16))) __bf16 v16bf;
typedef __attribute__((ext_vector_type(8)))  __bf16 v8bf;
typedef __attribute__((ext_vector_type(8)))  float  v8f;

__device__ __forceinline__ unsigned short f2bf(float f) {
    unsigned u = __float_as_uint(f);
    u += 0x7fffu + ((u >> 16) & 1u);      // round-to-nearest-even
    return (unsigned short)(u >> 16);
}
__device__ __forceinline__ float sigm(float x) {
    return 1.0f / (1.0f + __expf(-x));
}

// ---------------- prep: fold norm_w into gathered expert weights (bf16) ----------------
__global__ void mhc_prep_kernel(const long long* __restrict__ aidx,
                                const float* __restrict__ norm_w,
                                const float* __restrict__ pre_w,
                                const float* __restrict__ post_w,
                                const float* __restrict__ res_w,
                                const float* __restrict__ b_pre,
                                const float* __restrict__ b_post,
                                const float* __restrict__ b_res,
                                const float* __restrict__ a_pre,
                                const float* __restrict__ a_post,
                                const float* __restrict__ a_res,
                                unsigned short* __restrict__ wb,
                                float* __restrict__ params) {
    const int r = blockIdx.x;           // 0..191
    const int tid = threadIdx.x;
    const int k = r / 24, j = r % 24;
    const long long e = aidx[k];
    const float* src;
    if (j < 4)      src = pre_w  + ((size_t)e * NS + j) * NDH;
    else if (j < 8) src = post_w + ((size_t)e * NS + (j - 4)) * NDH;
    else            src = res_w  + ((size_t)e * 16 + (j - 8)) * NDH;
    const float* nw = norm_w + (size_t)e * NDH;
    for (int d = tid; d < NDH; d += 256)
        wb[(size_t)r * NDH + d] = f2bf(src[d] * nw[d]);

    if (r == 0 && tid < KACT * 28) {    // gathered scalars: [b_pre4 b_post4 b_res16 a_pre a_post a_res pad]
        const int kk = tid / 28, f = tid % 28;
        const long long ee = aidx[kk];
        float v = 0.f;
        if (f < 4)        v = b_pre[ee * NS + f];
        else if (f < 8)   v = b_post[ee * NS + (f - 4)];
        else if (f < 24)  v = b_res[ee * 16 + (f - 8)];
        else if (f == 24) v = a_pre[ee];
        else if (f == 25) v = a_post[ee];
        else if (f == 26) v = a_res[ee];
        params[kk * 28 + f] = v;
    }
}

// ---------------- main: load+rms -> WMMA GEMM -> gates + Sinkhorn ----------------
__launch_bounds__(256)
__global__ void mhc_main_kernel(const float* __restrict__ xin,
                                const unsigned short* __restrict__ wb,
                                const float* __restrict__ params,
                                float* __restrict__ out) {
    __shared__ unsigned char xs[MBLK * XS_STRIDE];   // bf16 raw activations, padded rows
    __shared__ unsigned char zs[MBLK * ZS_STRIDE];   // f32 GEMM results, padded rows
    __shared__ float rmsArr[MBLK];

    const int tid = threadIdx.x;
    const int w   = tid >> 5;        // wave 0..7
    const int l   = tid & 31;        // lane
    const int g0  = blockIdx.x * MBLK;
    const int b   = g0 >> 12;        // token g = b*T + t
    const int t0  = g0 & (TT - 1);

    // ---- Phase 1: read stream once; sum-of-squares; raw bf16 -> LDS ----
    for (int c = 0; c < 4; ++c) {
        const int tok = w * 4 + c;
        const int t = t0 + tok;
        float ss = 0.f;
        for (int n = 0; n < NS; ++n) {
            const float* sp = xin + (((size_t)(b * NS + n)) * TT + t) * DH;
            const unsigned kbase = n * DH;
            #pragma unroll
            for (int jj = 0; jj < 4; ++jj) {
                const float4 v = *(const float4*)(sp + jj * 128 + l * 4);
                ss = fmaf(v.x, v.x, fmaf(v.y, v.y, fmaf(v.z, v.z, fmaf(v.w, v.w, ss))));
                uint2 pk;
                pk.x = (unsigned)f2bf(v.x) | ((unsigned)f2bf(v.y) << 16);
                pk.y = (unsigned)f2bf(v.z) | ((unsigned)f2bf(v.w) << 16);
                *(uint2*)(xs + (size_t)tok * XS_STRIDE + (size_t)(kbase + jj * 128 + l * 4) * 2) = pk;
            }
        }
        for (int m = 16; m >= 1; m >>= 1) ss += __shfl_xor(ss, m, 32);
        if (l == 0) rmsArr[tok] = rsqrtf(ss * (1.0f / NDH) + 1e-8f);
    }
    __syncthreads();

    // ---- Phase 2: bf16 WMMA GEMM: z[tok][n] = x_raw . Wfused[n] ----
    // wave -> (mtile in {0,1}, 3 consecutive N-tiles of 12)
    const int mtile = w >> 2;
    const int q     = w & 3;
    const int h     = l >> 4;        // lane half
    const int lm    = l & 15;
    const unsigned aoff = (unsigned)(mtile * 16 + lm) * XS_STRIDE + h * 32;   // A: K = kc*32 + h*16 .. +15
    unsigned boff[3];
    #pragma unroll
    for (int nt = 0; nt < 3; ++nt)
        boff[nt] = (unsigned)(((q * 3 + nt) * 16 + lm) * (NDH * 2)) + h * 32; // B: K = kc*32 + h*16 .. +15

    v8f acc[3] = {};
    const unsigned char* wbb = (const unsigned char*)wb;
    for (int kc = 0; kc < 64; ++kc) {
        const v8bf alo = *(const v8bf*)(xs + aoff + kc * 64);
        const v8bf ahi = *(const v8bf*)(xs + aoff + kc * 64 + 16);
        const v16bf a = __builtin_shufflevector(alo, ahi,
            0, 1, 2, 3, 4, 5, 6, 7, 8, 9, 10, 11, 12, 13, 14, 15);
        if (kc + 1 < 64)
            __builtin_prefetch(wbb + boff[0] + (unsigned)(kc + 1) * 64, 0, 2);
        #pragma unroll
        for (int nt = 0; nt < 3; ++nt) {
            const v16bf bm = *(const v16bf*)(wbb + boff[nt] + (unsigned)kc * 64);
            acc[nt] = __builtin_amdgcn_wmma_f32_16x16x32_bf16(
                false, a, false, bm, (short)0, acc[nt], false, false);
        }
    }
    // C tile -> zs (VGPR i: rows M = i (lanes 0-15) / 8+i (lanes 16-31), N = lane&15)
    #pragma unroll
    for (int nt = 0; nt < 3; ++nt) {
        const int ncol = (q * 3 + nt) * 16 + lm;
        #pragma unroll
        for (int i = 0; i < 8; ++i) {
            const int tok = mtile * 16 + h * 8 + i;
            *(float*)(zs + (size_t)tok * ZS_STRIDE + ncol * 4) = acc[nt][i];
        }
    }
    __syncthreads();

    // ---- Phase 3: rms fold + gates + Sinkhorn; one (expert, token) per thread ----
    {
        const int k = w;           // 8 waves == 8 active experts
        const int tok = l;         // 32 tokens
        const float rms = rmsArr[tok];
        const float* P = params + k * 28;
        float z[24];
        #pragma unroll
        for (int j = 0; j < 24; ++j)
            z[j] = *(const float*)(zs + (size_t)tok * ZS_STRIDE + (k * 24 + j) * 4);

        const float apre = P[24] * rms, apost = P[25] * rms, ares = P[26] * rms;
        const size_t g = (size_t)g0 + tok;

        float4 pre, post;
        pre.x = sigm(apre * z[0] + P[0]);  pre.y = sigm(apre * z[1] + P[1]);
        pre.z = sigm(apre * z[2] + P[2]);  pre.w = sigm(apre * z[3] + P[3]);
        post.x = 2.f * sigm(apost * z[4] + P[4]);  post.y = 2.f * sigm(apost * z[5] + P[5]);
        post.z = 2.f * sigm(apost * z[6] + P[6]);  post.w = 2.f * sigm(apost * z[7] + P[7]);

        float m4[4][4];
        #pragma unroll
        for (int i = 0; i < 4; ++i)
            #pragma unroll
            for (int j = 0; j < 4; ++j)
                m4[i][j] = __expf(ares * z[8 + i * 4 + j] + P[8 + i * 4 + j]);
        #pragma unroll
        for (int it = 0; it < 6; ++it) {
            #pragma unroll
            for (int i = 0; i < 4; ++i) {           // rows (axis -1)
                const float inv = 1.0f / (m4[i][0] + m4[i][1] + m4[i][2] + m4[i][3]);
                m4[i][0] *= inv; m4[i][1] *= inv; m4[i][2] *= inv; m4[i][3] *= inv;
            }
            #pragma unroll
            for (int j = 0; j < 4; ++j) {           // cols (axis -2)
                const float inv = 1.0f / (m4[0][j] + m4[1][j] + m4[2][j] + m4[3][j]);
                m4[0][j] *= inv; m4[1][j] *= inv; m4[2][j] *= inv; m4[3][j] *= inv;
            }
        }

        float* outRes = out + ((size_t)k * NTOK + g) * 16;
        #pragma unroll
        for (int i = 0; i < 4; ++i) {
            float4 r4; r4.x = m4[i][0]; r4.y = m4[i][1]; r4.z = m4[i][2]; r4.w = m4[i][3];
            *(float4*)(outRes + i * 4) = r4;
        }
        float* outPre  = out + (size_t)KACT * NTOK * 16 + ((size_t)k * NTOK + g) * 4;
        float* outPost = out + (size_t)KACT * NTOK * 20 + ((size_t)k * NTOK + g) * 4;
        *(float4*)outPre  = pre;
        *(float4*)outPost = post;
    }
}

extern "C" void kernel_launch(void* const* d_in, const int* in_sizes, int n_in,
                              void* d_out, int out_size, void* d_ws, size_t ws_size,
                              hipStream_t hstream) {
    const float*     xin    = (const float*)d_in[0];
    const long long* aidx   = (const long long*)d_in[1];   // int64 active_idx
    const float*     norm_w = (const float*)d_in[2];
    const float*     pre_w  = (const float*)d_in[3];
    const float*     post_w = (const float*)d_in[4];
    const float*     res_w  = (const float*)d_in[5];
    const float*     b_pre  = (const float*)d_in[6];
    const float*     b_post = (const float*)d_in[7];
    const float*     b_res  = (const float*)d_in[8];
    const float*     a_pre  = (const float*)d_in[9];
    const float*     a_post = (const float*)d_in[10];
    const float*     a_res  = (const float*)d_in[11];

    unsigned short* wb     = (unsigned short*)d_ws;                       // 192*2048*2 = 768 KB
    float*          params = (float*)((char*)d_ws + (size_t)NROWS * NDH * 2);

    mhc_prep_kernel<<<NROWS, 256, 0, hstream>>>(aidx, norm_w, pre_w, post_w, res_w,
                                                b_pre, b_post, b_res,
                                                a_pre, a_post, a_res, wb, params);
    mhc_main_kernel<<<NTOK / MBLK, 256, 0, hstream>>>(xin, wb, params, (float*)d_out);
}
